// ModulatedSirenGridMix_26800595927126
// MI455X (gfx1250) — matrix operand order repaired and checked by reading, not compiled, not run on hardware
//
#include <hip/hip_runtime.h>

typedef __attribute__((ext_vector_type(16))) _Float16 v16h;
typedef __attribute__((ext_vector_type(8)))  _Float16 v8h;
typedef __attribute__((ext_vector_type(8)))  float    v8f;

namespace {
constexpr int   kB      = 8;
constexpr int   kP      = 16384;
constexpr int   kDH     = 256;      // hidden width
constexpr int   kGC     = 4;        // grid channels (layers with shifts)
constexpr int   kNGB    = 64;       // grid bases
constexpr int   kGS     = 32;       // grid spatial
constexpr int   kCells  = kGS * kGS;        // 1024
constexpr int   kMTile  = 128;      // points per workgroup
constexpr int   kThreads = 512;     // 16 wave32
constexpr int   kHStr   = kDH + 8;  // padded f16 row stride (528B: conflict-free b128 reads)
constexpr int   kWStr   = kDH + 8;
constexpr float kW0     = 30.0f;

constexpr size_t kGmElems = (size_t)kB * kGC * kCells * kDH;  // mixed grid, f32, layout (b,c,cell,d)

constexpr size_t kSmemBytes =
    (size_t)(kMTile * kHStr + kDH * kWStr) * sizeof(_Float16)   // lds_h + lds_w
    + (size_t)kMTile * 4 * sizeof(float)                        // corner weights
    + (size_t)kMTile * 4 * sizeof(int);                         // corner cell offsets

__device__ __forceinline__ int iclamp(int v, int lo, int hi) {
  return v < lo ? lo : (v > hi ? hi : v);
}

// CDNA5 async global->LDS copy, 16B per lane, tracked by ASYNCcnt.
// dsaddr = low 32 bits of the flat shared-memory address (LDS aperture rule).
__device__ __forceinline__ void async_load_b128_to_lds(const void* gptr, void* lptr) {
  unsigned int lds = (unsigned int)(unsigned long long)lptr;
  asm volatile("global_load_async_to_lds_b128 %0, %1, off"
               :
               : "v"(lds), "v"(gptr)
               : "memory");
}

__device__ __forceinline__ void wait_asynccnt0() {
  asm volatile("s_wait_asynccnt 0x0" ::: "memory");
}
} // namespace

// ---------------------------------------------------------------------------
// Kernel 1: fp32 -> f16 copies of W1, W2, W3 (65536 elements each).
// ---------------------------------------------------------------------------
__global__ void convert_weights_kernel(const float* __restrict__ W1,
                                       const float* __restrict__ W2,
                                       const float* __restrict__ W3,
                                       _Float16* __restrict__ wf16) {
  int i = blockIdx.x * 256 + threadIdx.x;             // grid = 256 blocks -> 65536 threads
  wf16[i]                 = (_Float16)W1[i];
  wf16[i + kDH * kDH]     = (_Float16)W2[i];
  wf16[i + 2 * kDH * kDH] = (_Float16)W3[i];
}

// ---------------------------------------------------------------------------
// Kernel 2: grid mixing. gm[b][c][cell][d] = sum_n (latent+bias)[b,c,n] * GB[c,n,d,cell]
// Block = one (c, d, 256-wide hw slab): grid_bases reads are fully coalesced
// (the 256MB stream — the HBM-bound phase, ~11us at 23.3 TB/s). Scattered 4B
// writes of the 33MB result are absorbed by the 192MB L2.
// ---------------------------------------------------------------------------
__global__ void grid_mix_kernel(const float* __restrict__ latent,
                                const float* __restrict__ grid_bases,
                                const float* __restrict__ mod_bias,
                                float* __restrict__ gm) {
  __shared__ float mods[kB * kNGB];
  const int blk  = blockIdx.x;          // 4 * 256 * 4 = 4096 blocks
  const int c    = blk >> 10;
  const int rem  = blk & 1023;
  const int d    = rem >> 2;
  const int hw   = ((rem & 3) << 8) + threadIdx.x;

  for (int i = threadIdx.x; i < kB * kNGB; i += 256) {
    int b = i >> 6, n = i & 63;
    mods[i] = latent[b * (kGC * kNGB) + c * kNGB + n] + mod_bias[c * kNGB + n];
  }
  __syncthreads();

  float acc[kB];
#pragma unroll
  for (int b = 0; b < kB; ++b) acc[b] = 0.f;

  const float* gb = grid_bases + ((size_t)(c * kNGB) * kDH + d) * kCells + hw;
  for (int n = 0; n < kNGB; ++n) {
    float g = gb[(size_t)n * kDH * kCells];
#pragma unroll
    for (int b = 0; b < kB; ++b) acc[b] = fmaf(mods[b * kNGB + n], g, acc[b]);
  }
#pragma unroll
  for (int b = 0; b < kB; ++b)
    gm[((size_t)(b * kGC + c) * kCells + hw) * kDH + d] = acc[b];
}

// ---------------------------------------------------------------------------
// Kernel 3: fused modulated-SIREN.
//   - 1024 blocks (8 batches x 128 point-tiles), 512 threads = 16 wave32.
//   - h tile (128x256 f16) and current-layer W (256x256 f16) staged in LDS.
//   - Weights arrive via GLOBAL_LOAD_ASYNC_TO_LDS_B128, issued one layer ahead
//     so the 128KB fetch is hidden under the epilogue's f32 shift gather
//     (layer-1 weights are prefetched under the layer-0 VALU work).
//   - Wave w owns a 32x64 output tile: 2 M-subtiles x 4 N-subtiles of 16x16,
//     accumulated with v_wmma_f32_16x16x32_f16 over K=256 (8 steps).
//   - f32 bilinear shifts gathered straight from the mixed grid in the epilogue
//     (keeps the x30-amplified shift path in full precision).
// CDNA5 fragment layouts used (CDNA5 ISA):
//   A(16x32 f16): lane m=L%16, half=L/16; v0..3 <- K = half*8 + [0..7],
//                 v4..7 <- K = 16 + half*8 + [0..7]   (two 16B LDS loads)
//   B(32x16 f16): lane n=L%16, half=L/16; v0..3 <- K = half*8 + [0..7], v4..7 <- K = 16 + half*8 + [0..7]
//                 (two 16B LDS loads), B[k][n] = W[n][k]
//   C/D(16x16 f32): element (M,N): vgpr = M%8, lane = (M/8)*16 + N
// ---------------------------------------------------------------------------
__device__ __forceinline__ void stage_weights_async(_Float16* lds_w,
                                                    const _Float16* wsrc, int tid) {
  // 8192 16B chunks / 512 threads = 16 async loads per thread (ASYNCcnt <= 16).
  for (int i = tid * 8; i < kDH * kDH; i += kThreads * 8) {
    int row = i >> 8, col = i & 255;
    async_load_b128_to_lds(wsrc + i, &lds_w[row * kWStr + col]);
  }
}

__launch_bounds__(kThreads, 1)
__global__ void siren_fused_kernel(const float* __restrict__ xg,
                                   const float* __restrict__ gm,
                                   const _Float16* __restrict__ wf16,
                                   const float* __restrict__ W0g,
                                   const float* __restrict__ b0g,
                                   const float* __restrict__ b1g,
                                   const float* __restrict__ b2g,
                                   const float* __restrict__ b3g,
                                   const float* __restrict__ Wlg,
                                   const float* __restrict__ blg,
                                   float* __restrict__ out) {
  extern __shared__ char smem[];
  _Float16* lds_h  = (_Float16*)smem;                    // [128][264]
  _Float16* lds_w  = lds_h + kMTile * kHStr;             // [256][264]
  float*    c_wt   = (float*)(lds_w + kDH * kWStr);      // [128][4]
  int*      c_cell = (int*)(c_wt + kMTile * 4);          // [128][4]

  const int tid = threadIdx.x;
  const int b   = blockIdx.x >> 7;
  const int p0  = (blockIdx.x & 127) * kMTile;

  // Prefetch layer-1 weights into LDS while we do the VALU-heavy setup below.
  stage_weights_async(lds_w, wf16, tid);

  // ---- per-point bilinear corners: coords = x*2-1 -> ix = x*31, iy = y*31 ----
  if (tid < kMTile) {
    const int p = tid;
    float px = xg[((size_t)b * kP + p0 + p) * 2 + 0];
    float py = xg[((size_t)b * kP + p0 + p) * 2 + 1];
    float ix = px * (float)(kGS - 1);
    float iy = py * (float)(kGS - 1);
    float fx = floorf(ix), fy = floorf(iy);
    float tx = ix - fx,   ty = iy - fy;
    c_wt[p * 4 + 0] = (1.f - tx) * (1.f - ty);
    c_wt[p * 4 + 1] = tx * (1.f - ty);
    c_wt[p * 4 + 2] = (1.f - tx) * ty;
    c_wt[p * 4 + 3] = tx * ty;
    int x0 = iclamp((int)fx,     0, kGS - 1);
    int x1 = iclamp((int)fx + 1, 0, kGS - 1);
    int y0 = iclamp((int)fy,     0, kGS - 1);
    int y1 = iclamp((int)fy + 1, 0, kGS - 1);
    c_cell[p * 4 + 0] = y0 * kGS + x0;
    c_cell[p * 4 + 1] = y0 * kGS + x1;
    c_cell[p * 4 + 2] = y1 * kGS + x0;
    c_cell[p * 4 + 3] = y1 * kGS + x1;
  }
  __syncthreads();

  // ---- layer 0 (K=2, plain VALU): h0 = sin(30*(x@W0^T + b0 + shift0)) ----
  {
    const float* gmb = gm + (size_t)(b * kGC + 0) * kCells * kDH;
    for (int idx = tid; idx < kMTile * kDH; idx += kThreads) {
      const int p = idx >> 8, d = idx & 255;
      const float* wt = &c_wt[p * 4];
      const int*   cl = &c_cell[p * 4];
      float s = wt[0] * gmb[cl[0] * kDH + d] + wt[1] * gmb[cl[1] * kDH + d]
              + wt[2] * gmb[cl[2] * kDH + d] + wt[3] * gmb[cl[3] * kDH + d];
      float px = xg[((size_t)b * kP + p0 + p) * 2 + 0];
      float py = xg[((size_t)b * kP + p0 + p) * 2 + 1];
      float v = fmaf(px, W0g[d * 2 + 0], fmaf(py, W0g[d * 2 + 1], b0g[d] + s));
      lds_h[p * kHStr + d] = (_Float16)__sinf(kW0 * v);
    }
  }

  // ---- wave tiling for the WMMA layers ----
  const int wv     = tid >> 5;
  const int lane   = tid & 31;
  const int l16    = lane & 15;    // A: row-in-Mtile; B/C: col-in-Ntile
  const int half   = lane >> 4;
  const int mgroup = wv & 3;       // M rows mgroup*32 .. +31
  const int ngroup = wv >> 2;      // N cols ngroup*64 .. +63

  for (int layer = 1; layer <= 3; ++layer) {
    // Publish lds_w (this layer's async weight loads) and lds_h (previous
    // activation writes): each thread drains its own ASYNCcnt, the barrier
    // makes everything visible workgroup-wide.
    wait_asynccnt0();
    __syncthreads();

    // ---- GEMM: C(32x64 per wave) = h(128x256) @ W^T, K in 8 steps of 32 ----
    v8f C[2][4] = {};
    for (int kk = 0; kk < 8; ++kk) {
      const int kb = kk * 32;
      v16h A[2], Bm[4];
#pragma unroll
      for (int mt = 0; mt < 2; ++mt) {
        const _Float16* pa = &lds_h[(mgroup * 32 + mt * 16 + l16) * kHStr + kb + half * 8];
        v8h lo = *(const v8h*)pa;
        v8h hi = *(const v8h*)(pa + 16);
        A[mt] = __builtin_shufflevector(lo, hi, 0,1,2,3,4,5,6,7,8,9,10,11,12,13,14,15);
      }
#pragma unroll
      for (int nt = 0; nt < 4; ++nt) {
        // B[k][n] = W[n][k]
        const _Float16* pb = &lds_w[(ngroup * 64 + nt * 16 + l16) * kWStr + kb + half * 8];
        v8h lo = *(const v8h*)pb;
        v8h hi = *(const v8h*)(pb + 16);
        Bm[nt] = __builtin_shufflevector(lo, hi, 0,1,2,3,4,5,6,7,8,9,10,11,12,13,14,15);
      }
#pragma unroll
      for (int mt = 0; mt < 2; ++mt)
#pragma unroll
        for (int nt = 0; nt < 4; ++nt)
          C[mt][nt] = __builtin_amdgcn_wmma_f32_16x16x32_f16(
              false, A[mt], false, Bm[nt], (short)0, C[mt][nt], false, false);
    }
    __syncthreads();   // all waves done reading lds_h and lds_w

    // lds_w is now dead until the next iteration's barrier: prefetch the next
    // layer's weights asynchronously underneath the gather-heavy epilogue.
    if (layer < 3)
      stage_weights_async(lds_w, wf16 + (size_t)layer * kDH * kDH, tid);

    // ---- epilogue: + bias + f32 bilinear shift, sin(30*.), back to LDS f16 ----
    const float* bg  = (layer == 1) ? b1g : (layer == 2) ? b2g : b3g;
    const float* gmb = gm + (size_t)(b * kGC + layer) * kCells * kDH;
#pragma unroll
    for (int nt = 0; nt < 4; ++nt) {
      const int n = ngroup * 64 + nt * 16 + l16;
      const float bias = bg[n];
#pragma unroll
      for (int mt = 0; mt < 2; ++mt) {
#pragma unroll
        for (int v = 0; v < 8; ++v) {
          const int m = mgroup * 32 + mt * 16 + half * 8 + v;
          const float* wt = &c_wt[m * 4];
          const int*   cl = &c_cell[m * 4];
          float s = wt[0] * gmb[cl[0] * kDH + n] + wt[1] * gmb[cl[1] * kDH + n]
                  + wt[2] * gmb[cl[2] * kDH + n] + wt[3] * gmb[cl[3] * kDH + n];
          float val = C[mt][nt][v] + bias + s;
          lds_h[m * kHStr + n] = (_Float16)__sinf(kW0 * val);
        }
      }
    }
  }
  __syncthreads();

  // ---- final layer: 256 -> 3, no activation ----
  for (int idx = tid; idx < kMTile * 3; idx += kThreads) {
    const int p = idx / 3, o = idx - p * 3;
    float acc = blg[o];
    const float* wl = Wlg + o * kDH;
    for (int i = 0; i < kDH; ++i)
      acc = fmaf((float)lds_h[p * kHStr + i], wl[i], acc);
    out[((size_t)b * kP + p0 + p) * 3 + o] = acc;
  }
}

// ---------------------------------------------------------------------------
extern "C" void kernel_launch(void* const* d_in, const int* in_sizes, int n_in,
                              void* d_out, int out_size, void* d_ws, size_t ws_size,
                              hipStream_t stream) {
  (void)in_sizes; (void)n_in; (void)out_size; (void)ws_size;

  const float* x          = (const float*)d_in[0];
  const float* latent     = (const float*)d_in[1];
  const float* grid_bases = (const float*)d_in[2];
  const float* mod_bias   = (const float*)d_in[3];
  const float* W0         = (const float*)d_in[4];
  const float* b0         = (const float*)d_in[5];
  const float* W1         = (const float*)d_in[6];
  const float* b1         = (const float*)d_in[7];
  const float* W2         = (const float*)d_in[8];
  const float* b2         = (const float*)d_in[9];
  const float* W3         = (const float*)d_in[10];
  const float* b3         = (const float*)d_in[11];
  const float* Wl         = (const float*)d_in[12];
  const float* bl         = (const float*)d_in[13];
  float*       out        = (float*)d_out;

  // workspace: [ gm f32 : 33.5MB ][ wf16 : 384KB ]   (needs ~34MB)
  float*    gm   = (float*)d_ws;
  _Float16* wf16 = (_Float16*)(gm + kGmElems);

  (void)hipFuncSetAttribute((const void*)siren_fused_kernel,
                            hipFuncAttributeMaxDynamicSharedMemorySize,
                            (int)kSmemBytes);

  convert_weights_kernel<<<256, 256, 0, stream>>>(W1, W2, W3, wf16);
  grid_mix_kernel<<<kGC * kDH * 4, 256, 0, stream>>>(latent, grid_bases, mod_bias, gm);
  siren_fused_kernel<<<kB * (kP / kMTile), kThreads, kSmemBytes, stream>>>(
      x, gm, wf16, W0, b0, b1, b2, b3, Wl, bl, out);
}
